// WeightedGIN_74955769249853
// MI455X (gfx1250) — compile-verified, matrix operand order checked by reading, not run on hardware
//
#include <hip/hip_runtime.h>

#define N_NODES 50000
#define N_EDGES 800000
#define D 256

typedef unsigned short u16;
typedef unsigned int   u32;
typedef __attribute__((ext_vector_type(16))) __bf16 v16bf;
typedef __attribute__((ext_vector_type(8)))  float  v8f;
typedef __attribute__((ext_vector_type(4)))  u32    u32x4;

union Frag16 { u32x4 q[2]; v16bf v; };

// f32 -> bf16 bits, round-to-nearest-even (bit manipulation only; no __bf16 math)
__device__ __forceinline__ u16 f2bf(float f) {
  u32 b = __float_as_uint(f);
  b += 0x7FFFu + ((b >> 16) & 1u);
  return (u16)(b >> 16);
}

// ---------------------------------------------------------------- weights prep
// WT[n][k] = bf16(W[k][n])  (row-major [256][256])
__global__ void __launch_bounds__(256) k_prep_weights(const float* __restrict__ W1,
                                                      const float* __restrict__ W2,
                                                      u16* __restrict__ W1T,
                                                      u16* __restrict__ W2T) {
  int n = blockIdx.x, k = threadIdx.x;
  W1T[n * D + k] = f2bf(W1[k * D + n]);
  W2T[n * D + k] = f2bf(W2[k * D + n]);
}

__global__ void k_zero_stats(float* __restrict__ stats) {
  stats[threadIdx.x] = 0.0f;  // 512 threads: sum[256] + sumsq[256]
}

// ---------------------------------------------------------------- h = (1+eps)*x
__global__ void __launch_bounds__(256) k_init_h(const float* __restrict__ x,
                                                const float* __restrict__ epsp,
                                                float* __restrict__ h) {
  size_t i = (size_t)blockIdx.x * 256 + threadIdx.x;
  float s = 1.0f + epsp[0];
  float4 v = reinterpret_cast<const float4*>(x)[i];
  v.x *= s; v.y *= s; v.z *= s; v.w *= s;
  reinterpret_cast<float4*>(h)[i] = v;
}

// ---------------------------------------------------------------- edge scatter
// one wave (32 lanes) per edge; 8 floats/lane; hardware f32 atomics (L2-resident)
__global__ void __launch_bounds__(256) k_scatter(const float* __restrict__ x,
                                                 const long long* __restrict__ ei,
                                                 const float* __restrict__ ew,
                                                 float* __restrict__ h) {
  int lane = threadIdx.x & 31;
  int wv   = threadIdx.x >> 5;
  long long e = (long long)blockIdx.x * 8 + wv;
  if (e >= N_EDGES) return;
  long long s = ei[e];
  long long d = ei[N_EDGES + e];
  float w = ew[e];
  const float4* xs = reinterpret_cast<const float4*>(x + (size_t)s * D);
  float* hd = h + (size_t)d * D;
#pragma unroll
  for (int c = 0; c < 2; ++c) {
    float4 v = xs[c * 32 + lane];     // 512B contiguous per wave per chunk
    int base = c * 128 + lane * 4;
    unsafeAtomicAdd(hd + base + 0, v.x * w);
    unsafeAtomicAdd(hd + base + 1, v.y * w);
    unsafeAtomicAdd(hd + base + 2, v.z * w);
    unsafeAtomicAdd(hd + base + 3, v.w * w);
  }
}

// ---------------------------------------------------------------- GEMM1: h1 = h@W1 + b1
// block: 16 rows x 256 cols; 8 waves, 2 col-tiles each; bf16 WMMA, f32 accum
__global__ void __launch_bounds__(256) k_gemm1(const float* __restrict__ h,
                                               const u16* __restrict__ WT,
                                               const float* __restrict__ bias,
                                               float* __restrict__ h1) {
  __shared__ alignas(16) u16 As[16 * 264];   // padded stride: bank-conflict free
  const int t = threadIdx.x;
  const int m0 = blockIdx.x * 16;
  {
    int r  = t >> 4;
    int ks = (t & 15) * 16;
    const float4* src = reinterpret_cast<const float4*>(h + (size_t)(m0 + r) * D + ks);
    u32 packed[8];
#pragma unroll
    for (int j = 0; j < 4; ++j) {
      float4 v = src[j];
      packed[j * 2 + 0] = (u32)f2bf(v.x) | ((u32)f2bf(v.y) << 16);
      packed[j * 2 + 1] = (u32)f2bf(v.z) | ((u32)f2bf(v.w) << 16);
    }
    u32* dst = reinterpret_cast<u32*>(&As[r * 264 + ks]);
#pragma unroll
    for (int j = 0; j < 8; ++j) dst[j] = packed[j];
  }
  __syncthreads();

  const int lane = t & 31;
  const int wv   = t >> 5;
  const int m    = lane & 15;
  const int kh   = (lane >> 4) * 8;        // 16-bit WMMA lane K-offset
  const int n0   = wv * 32 + m;            // this lane's column in tile 0
  v8f c0 = {}; v8f c1 = {};
#pragma unroll
  for (int kb = 0; kb < 8; ++kb) {
    int kbase = kb * 32 + kh;
    Frag16 a, b0, b1f;
    const u16* ap = &As[m * 264 + kbase];
    a.q[0] = *reinterpret_cast<const u32x4*>(ap);
    a.q[1] = *reinterpret_cast<const u32x4*>(ap + 16);
    const u16* bp0 = WT + (size_t)n0 * D + kbase;
    b0.q[0]  = *reinterpret_cast<const u32x4*>(bp0);
    b0.q[1]  = *reinterpret_cast<const u32x4*>(bp0 + 16);
    const u16* bp1 = bp0 + 16 * D;
    b1f.q[0] = *reinterpret_cast<const u32x4*>(bp1);
    b1f.q[1] = *reinterpret_cast<const u32x4*>(bp1 + 16);
    c0 = __builtin_amdgcn_wmma_f32_16x16x32_bf16(false, a.v, false, b0.v,  (short)0, c0, false, false);
    c1 = __builtin_amdgcn_wmma_f32_16x16x32_bf16(false, a.v, false, b1f.v, (short)0, c1, false, false);
  }
  const int col0 = n0, col1 = n0 + 16;
  const float bb0 = bias[col0], bb1 = bias[col1];
#pragma unroll
  for (int vv = 0; vv < 8; ++vv) {
    size_t row = (size_t)(m0 + vv + ((lane >> 4) * 8));
    h1[row * D + col0] = c0[vv] + bb0;
    h1[row * D + col1] = c1[vv] + bb1;
  }
}

// ---------------------------------------------------------------- column stats
__global__ void __launch_bounds__(256) k_colstats(const float* __restrict__ h1,
                                                  float* __restrict__ sum,
                                                  float* __restrict__ sumsq) {
  int col = threadIdx.x;
  size_t r0 = (size_t)blockIdx.x * 100;
  float s = 0.0f, q = 0.0f;
  for (int i = 0; i < 100; ++i) {
    float v = h1[(r0 + i) * D + col];
    s += v; q += v * v;
  }
  unsafeAtomicAdd(&sum[col], s);
  unsafeAtomicAdd(&sumsq[col], q);
}

__global__ void k_finalize(const float* __restrict__ sum, const float* __restrict__ sumsq,
                           const float* __restrict__ gamma, const float* __restrict__ beta,
                           float* __restrict__ scale, float* __restrict__ shift) {
  int c = threadIdx.x;
  const float invN = 1.0f / (float)N_NODES;
  float mean = sum[c] * invN;
  float var  = sumsq[c] * invN - mean * mean;
  float inv  = rsqrtf(var + 1e-5f);
  float sc   = gamma[c] * inv;
  scale[c] = sc;
  shift[c] = beta[c] - mean * sc;
}

// ---------------------------------------------------------------- GEMM2: out = relu(norm(h1))@W2 + b2 + x
// in-place safe: each block only reads the rows it writes, all A reads precede writes
__global__ void __launch_bounds__(256) k_gemm2(const float* __restrict__ h1,
                                               const u16* __restrict__ WT,
                                               const float* __restrict__ bias,
                                               const float* __restrict__ x,
                                               const float* __restrict__ scale,
                                               const float* __restrict__ shift,
                                               float* __restrict__ out) {
  __shared__ alignas(16) u16 As[16 * 264];
  const int t = threadIdx.x;
  const int m0 = blockIdx.x * 16;
  {
    int r  = t >> 4;
    int ks = (t & 15) * 16;
    const float4* src = reinterpret_cast<const float4*>(h1 + (size_t)(m0 + r) * D + ks);
    u32 packed[8];
#pragma unroll
    for (int j = 0; j < 4; ++j) {
      float4 v  = src[j];
      float4 sc = reinterpret_cast<const float4*>(scale)[(ks >> 2) + j];
      float4 sh = reinterpret_cast<const float4*>(shift)[(ks >> 2) + j];
      float e0 = fmaxf(fmaf(v.x, sc.x, sh.x), 0.0f);
      float e1 = fmaxf(fmaf(v.y, sc.y, sh.y), 0.0f);
      float e2 = fmaxf(fmaf(v.z, sc.z, sh.z), 0.0f);
      float e3 = fmaxf(fmaf(v.w, sc.w, sh.w), 0.0f);
      packed[j * 2 + 0] = (u32)f2bf(e0) | ((u32)f2bf(e1) << 16);
      packed[j * 2 + 1] = (u32)f2bf(e2) | ((u32)f2bf(e3) << 16);
    }
    u32* dst = reinterpret_cast<u32*>(&As[r * 264 + ks]);
#pragma unroll
    for (int j = 0; j < 8; ++j) dst[j] = packed[j];
  }
  __syncthreads();

  const int lane = t & 31;
  const int wv   = t >> 5;
  const int m    = lane & 15;
  const int kh   = (lane >> 4) * 8;
  const int n0   = wv * 32 + m;
  v8f c0 = {}; v8f c1 = {};
#pragma unroll
  for (int kb = 0; kb < 8; ++kb) {
    int kbase = kb * 32 + kh;
    Frag16 a, b0, b1f;
    const u16* ap = &As[m * 264 + kbase];
    a.q[0] = *reinterpret_cast<const u32x4*>(ap);
    a.q[1] = *reinterpret_cast<const u32x4*>(ap + 16);
    const u16* bp0 = WT + (size_t)n0 * D + kbase;
    b0.q[0]  = *reinterpret_cast<const u32x4*>(bp0);
    b0.q[1]  = *reinterpret_cast<const u32x4*>(bp0 + 16);
    const u16* bp1 = bp0 + 16 * D;
    b1f.q[0] = *reinterpret_cast<const u32x4*>(bp1);
    b1f.q[1] = *reinterpret_cast<const u32x4*>(bp1 + 16);
    c0 = __builtin_amdgcn_wmma_f32_16x16x32_bf16(false, a.v, false, b0.v,  (short)0, c0, false, false);
    c1 = __builtin_amdgcn_wmma_f32_16x16x32_bf16(false, a.v, false, b1f.v, (short)0, c1, false, false);
  }
  const int col0 = n0, col1 = n0 + 16;
  const float bb0 = bias[col0], bb1 = bias[col1];
#pragma unroll
  for (int vv = 0; vv < 8; ++vv) {
    size_t row = (size_t)(m0 + vv + ((lane >> 4) * 8));
    out[row * D + col0] = c0[vv] + bb0 + x[row * D + col0];
    out[row * D + col1] = c1[vv] + bb1 + x[row * D + col1];
  }
}

// ---------------------------------------------------------------- launch
extern "C" void kernel_launch(void* const* d_in, const int* in_sizes, int n_in,
                              void* d_out, int out_size, void* d_ws, size_t ws_size,
                              hipStream_t stream) {
  const float*     x     = (const float*)d_in[0];
  const long long* ei    = (const long long*)d_in[1];
  const float*     ew    = (const float*)d_in[2];
  const float*     eps   = (const float*)d_in[3];
  const float*     W1    = (const float*)d_in[4];
  const float*     b1    = (const float*)d_in[5];
  const float*     gamma = (const float*)d_in[6];
  const float*     beta  = (const float*)d_in[7];
  const float*     W2    = (const float*)d_in[8];
  const float*     b2    = (const float*)d_in[9];
  float* out = (float*)d_out;

  // workspace layout (~51.5 MB): h | W1T(bf16) | W2T(bf16) | stats
  char* ws = (char*)d_ws;
  float* h     = (float*)ws;                        // 50000*256*4 = 51,200,000 B
  u16*   W1T   = (u16*)(ws + 51200000);             // 131,072 B
  u16*   W2T   = (u16*)(ws + 51331072);             // 131,072 B
  float* stats = (float*)(ws + 51462144);           // 1024 floats
  float* sum   = stats;
  float* sumsq = stats + 256;
  float* scale = stats + 512;
  float* shift = stats + 768;
  float* h1 = out;  // reuse output buffer for intermediate h1 (in-place safe)

  k_prep_weights<<<256, 256, 0, stream>>>(W1, W2, W1T, W2T);
  k_zero_stats  <<<1, 512, 0, stream>>>(stats);
  k_init_h      <<<(N_NODES * D / 4) / 256, 256, 0, stream>>>(x, eps, h);
  k_scatter     <<<N_EDGES / 8, 256, 0, stream>>>(x, ei, ew, h);
  k_gemm1       <<<N_NODES / 16, 256, 0, stream>>>(h, W1T, b1, h1);
  k_colstats    <<<N_NODES / 100, 256, 0, stream>>>(h1, sum, sumsq);
  k_finalize    <<<1, 256, 0, stream>>>(sum, sumsq, gamma, beta, scale, shift);
  k_gemm2       <<<N_NODES / 16, 256, 0, stream>>>(h1, W2T, b2, x, scale, shift, out);
}
